// HeteroGATEncoder_78984448573533
// MI455X (gfx1250) — matrix operand order repaired and checked by reading, not compiled, not run on hardware
//
#include <hip/hip_runtime.h>

// ---------------------------------------------------------------------------
// HeteroGAT encoder for MI455X (gfx1250).
// Dense GEMMs run on v_wmma_f32_16x16x32_bf16 (f32 accumulate) with weights
// pre-packed into per-lane WMMA B fragments; each wave feeds one A fragment
// into two WMMAs. Edge phase = 3-pass segment softmax with global float
// atomics (destination tables are L2-resident on the 192 MB L2).
// ---------------------------------------------------------------------------

#define DEVFN __device__ __forceinline__

typedef __attribute__((ext_vector_type(16))) __bf16 bf16x16;
typedef __attribute__((ext_vector_type(8)))  float  floatx8;

union BFU { bf16x16 v; unsigned short u[16]; };

DEVFN unsigned short f2bf(float x) {
  unsigned u = __float_as_uint(x);
  u += 0x7FFFu + ((u >> 16) & 1u);          // round-to-nearest-even
  return (unsigned short)(u >> 16);
}

DEVFN floatx8 wmma_bf16(bf16x16 a, bf16x16 b, floatx8 c) {
  return __builtin_amdgcn_wmma_f32_16x16x32_bf16(
      /*neg_a=*/false, a, /*neg_b=*/false, b,
      /*c_mod=*/(short)0, c, /*reuse_a=*/false, /*reuse_b=*/false);
}

// A fragment (16x32 bf16): lane L holds row m = m0 + (L&15); lane-half selects
// K base (+0 / +8); elements 0..7 <-> K+0..7, elements 8..15 <-> K+16..23.
DEVFN bf16x16 load_a_frag(const float* __restrict__ A, int K, int M,
                          int m0, int k0, int lane) {
  int m = m0 + (lane & 15);
  if (m >= M) m = M - 1;                    // clamp (tiles are exact here)
  const float* row = A + (size_t)m * K + (k0 + 8 * (lane >> 4));
  float4 f0 = *(const float4*)(row);
  float4 f1 = *(const float4*)(row + 4);
  float4 f2 = *(const float4*)(row + 16);
  float4 f3 = *(const float4*)(row + 20);
  BFU r;
  r.u[0]  = f2bf(f0.x); r.u[1]  = f2bf(f0.y); r.u[2]  = f2bf(f0.z); r.u[3]  = f2bf(f0.w);
  r.u[4]  = f2bf(f1.x); r.u[5]  = f2bf(f1.y); r.u[6]  = f2bf(f1.z); r.u[7]  = f2bf(f1.w);
  r.u[8]  = f2bf(f2.x); r.u[9]  = f2bf(f2.y); r.u[10] = f2bf(f2.z); r.u[11] = f2bf(f2.w);
  r.u[12] = f2bf(f3.x); r.u[13] = f2bf(f3.y); r.u[14] = f2bf(f3.z); r.u[15] = f2bf(f3.w);
  return r.v;
}

// ---------------------------------------------------------------------------
// Weight packing: W[K,N] f32 row-major -> bf16 fragments
//   P[((nt*KS + ks)*32 + lane)*16 + e] = bf16(W[(ks*32 + 8*(lane>>4) + (e&7) + 16*(e>>3))*N
//                                               + nt*16 + (lane&15)])
// ---------------------------------------------------------------------------
__global__ void pack_w_kernel(const float* __restrict__ W,
                              unsigned short* __restrict__ P, int K, int N) {
  int idx = blockIdx.x * 256 + threadIdx.x;
  int KS = K >> 5, NT = N >> 4;
  int total = NT * KS * 512;
  if (idx >= total) return;
  int e    = idx & 15;
  int lane = (idx >> 4) & 31;
  int rest = idx >> 9;
  int ks   = rest % KS;
  int nt   = rest / KS;
  int k = ks * 32 + 8 * (lane >> 4) + (e & 7) + 16 * (e >> 3);
  int n = nt * 16 + (lane & 15);
  P[idx] = f2bf(W[(size_t)k * N + n]);
}

__global__ void fill_u32_kernel(unsigned* __restrict__ p, unsigned v, size_t n) {
  size_t t = (size_t)blockIdx.x * 256 + threadIdx.x;
  if (t < n) p[t] = v;
}

// ---------------------------------------------------------------------------
// Generic GEMM + bias: C[M,N] = A[M,K] @ W + bias. Block = 8 waves; each wave
// owns TWO adjacent 16x16 tiles (one A fragment -> two WMMAs). Block covers
// 16 rows x 256 cols. grid = (ceil(M/16), N/256).
// ---------------------------------------------------------------------------
__global__ void gemm_bias_kernel(const float* __restrict__ A,
                                 const unsigned short* __restrict__ PW,
                                 const float* __restrict__ bias,
                                 float* __restrict__ C, int M, int N, int K) {
  int wave = threadIdx.x >> 5, lane = threadIdx.x & 31;
  int m0 = blockIdx.x * 16;
  int nt0 = (blockIdx.y * 8 + wave) * 2;    // first of two N-tiles
  int KS = K >> 5;
  floatx8 acc0 = {}, acc1 = {};
  const unsigned short* pw0 = PW + (size_t)nt0 * KS * 512 + lane * 16;
  const unsigned short* pw1 = pw0 + (size_t)KS * 512;
  for (int ks = 0; ks < KS; ks++) {
    bf16x16 a  = load_a_frag(A, K, M, m0, ks * 32, lane);
    bf16x16 b0 = *(const bf16x16*)(pw0 + (size_t)ks * 512);
    bf16x16 b1 = *(const bf16x16*)(pw1 + (size_t)ks * 512);
    acc0 = wmma_bf16(a, b0, acc0);
    acc1 = wmma_bf16(a, b1, acc1);
  }
  int nl = lane & 15, half = lane >> 4;
  int n0 = nt0 * 16 + nl;
  int n1 = n0 + 16;
  float bv0 = bias[n0], bv1 = bias[n1];
  int mbase = m0 + 8 * half;
  float* cp = C + (size_t)mbase * N;
  if (mbase + 8 <= M) {                     // fast path: full tile, no guards
#pragma unroll
    for (int r = 0; r < 8; r++) {
      cp[r * N + n0] = acc0[r] + bv0;
      cp[r * N + n1] = acc1[r] + bv1;
    }
  } else {
#pragma unroll
    for (int r = 0; r < 8; r++) {
      if (mbase + r < M) {
        cp[r * N + n0] = acc0[r] + bv0;
        cp[r * N + n1] = acc1[r] + bv1;
      }
    }
  }
}

// ---------------------------------------------------------------------------
// Fused GEMM(N=256) + bias + LayerNorm + ELU. Block = 8 waves, wave owns two
// 16x16 tiles -> full 16x256 row tile in LDS, then per-row LN via shuffles.
// ---------------------------------------------------------------------------
__global__ void gemm_ln_elu_kernel(const float* __restrict__ A,
                                   const unsigned short* __restrict__ PW,
                                   const float* __restrict__ bias,
                                   const float* __restrict__ gam,
                                   const float* __restrict__ bet,
                                   float* __restrict__ Out, int M, int K) {
  __shared__ float smem[16][264];
  int wave = threadIdx.x >> 5, lane = threadIdx.x & 31;
  int m0 = blockIdx.x * 16;
  int KS = K >> 5;
  int nt0 = wave * 2, nt1 = wave * 2 + 1;
  floatx8 acc0 = {}, acc1 = {};
  const unsigned short* pw0 = PW + (size_t)nt0 * KS * 512 + lane * 16;
  const unsigned short* pw1 = pw0 + (size_t)KS * 512;
  for (int ks = 0; ks < KS; ks++) {
    bf16x16 a  = load_a_frag(A, K, M, m0, ks * 32, lane);
    bf16x16 b0 = *(const bf16x16*)(pw0 + (size_t)ks * 512);
    bf16x16 b1 = *(const bf16x16*)(pw1 + (size_t)ks * 512);
    acc0 = wmma_bf16(a, b0, acc0);
    acc1 = wmma_bf16(a, b1, acc1);
  }
  int half = lane >> 4, nl = lane & 15;
#pragma unroll
  for (int r = 0; r < 8; r++) {
    int m = r + 8 * half;
    smem[m][nt0 * 16 + nl] = acc0[r] + bias[nt0 * 16 + nl];
    smem[m][nt1 * 16 + nl] = acc1[r] + bias[nt1 * 16 + nl];
  }
  __syncthreads();
#pragma unroll
  for (int rr = 0; rr < 2; rr++) {
    int row = wave * 2 + rr;
    if (m0 + row >= M) continue;
    float vals[8], s = 0.f, ss = 0.f;
#pragma unroll
    for (int j = 0; j < 8; j++) {
      float x = smem[row][lane + 32 * j];
      vals[j] = x; s += x; ss += x * x;
    }
#pragma unroll
    for (int msk = 16; msk >= 1; msk >>= 1) {
      s  += __shfl_xor(s,  msk, 32);
      ss += __shfl_xor(ss, msk, 32);
    }
    float mean = s * (1.f / 256.f);
    float var  = ss * (1.f / 256.f) - mean * mean;
    float inv  = rsqrtf(var + 1e-5f);
#pragma unroll
    for (int j = 0; j < 8; j++) {
      int c = lane + 32 * j;
      float y = (vals[j] - mean) * inv * gam[c] + bet[c];
      y = y > 0.f ? y : (__expf(y) - 1.f);
      Out[(size_t)(m0 + row) * 256 + c] = y;
    }
  }
}

// ---------------------------------------------------------------------------
// (x + bias) -> LayerNorm -> ELU, N=256, one wave per row.
// ---------------------------------------------------------------------------
__global__ void bias_ln_elu_kernel(const float* __restrict__ X,
                                   const float* __restrict__ bias,
                                   const float* __restrict__ gam,
                                   const float* __restrict__ bet,
                                   float* __restrict__ Out, int M) {
  int wave = threadIdx.x >> 5, lane = threadIdx.x & 31;
  int row = blockIdx.x * 8 + wave;
  if (row >= M) return;
  const float* x = X + (size_t)row * 256;
  float vals[8], s = 0.f, ss = 0.f;
#pragma unroll
  for (int j = 0; j < 8; j++) {
    int c = lane + 32 * j;
    float v = x[c] + bias[c];
    vals[j] = v; s += v; ss += v * v;
  }
#pragma unroll
  for (int msk = 16; msk >= 1; msk >>= 1) {
    s  += __shfl_xor(s,  msk, 32);
    ss += __shfl_xor(ss, msk, 32);
  }
  float mean = s * (1.f / 256.f);
  float var  = ss * (1.f / 256.f) - mean * mean;
  float inv  = rsqrtf(var + 1e-5f);
#pragma unroll
  for (int j = 0; j < 8; j++) {
    int c = lane + 32 * j;
    float y = (vals[j] - mean) * inv * gam[c] + bet[c];
    y = y > 0.f ? y : (__expf(y) - 1.f);
    Out[(size_t)row * 256 + c] = y;
  }
}

// ---------------------------------------------------------------------------
// Edge pass 1: logits[e,h] = att_h . leaky_relu(xl[src] + xr[dst]), plus a
// float segment-max via int/uint atomic trick (maxbuf preset to 0xFF800000).
// One wave per edge, 8 lanes per head.
// ---------------------------------------------------------------------------
template <int C>
__global__ void edge_logits_kernel(const int* __restrict__ srcI,
                                   const int* __restrict__ dstI,
                                   const float* __restrict__ xl,
                                   const float* __restrict__ xr,
                                   const float* __restrict__ att,
                                   float* __restrict__ logits,
                                   int* __restrict__ maxbuf, int E) {
  const int HC = 4 * C, PER = HC / 32;
  int wave = threadIdx.x >> 5, lane = threadIdx.x & 31;
  int e = blockIdx.x * 8 + wave;
  if (e >= E) return;
  int s = srcI[e], d = dstI[e];
  const float* pl = xl + (size_t)s * HC + lane * PER;
  const float* pr = xr + (size_t)d * HC + lane * PER;
  const float* pa = att + lane * PER;
  float sum = 0.f;
#pragma unroll
  for (int j = 0; j < PER; j++) {
    float v = pl[j] + pr[j];
    v = v > 0.f ? v : 0.2f * v;
    sum += pa[j] * v;
  }
  sum += __shfl_xor(sum, 1, 32);
  sum += __shfl_xor(sum, 2, 32);
  sum += __shfl_xor(sum, 4, 32);
  if ((lane & 7) == 0) {
    int h = lane >> 3;
    logits[(size_t)e * 4 + h] = sum;
    int idx = d * 4 + h;
    if (sum >= 0.f) atomicMax((int*)maxbuf + idx, __float_as_int(sum));
    else            atomicMin((unsigned*)maxbuf + idx, __float_as_uint(sum));
  }
}

// Edge pass 2: ex = exp(logit - max[dst]); store ex over logits; atomic sum.
__global__ void edge_exp_kernel(const int* __restrict__ dstI,
                                float* __restrict__ logits,
                                const int* __restrict__ maxbuf,
                                float* __restrict__ sumbuf, int E) {
  int t = blockIdx.x * 256 + threadIdx.x;
  if (t >= E * 4) return;
  int e = t >> 2, h = t & 3;
  int d = dstI[e];
  float mx = __int_as_float(maxbuf[d * 4 + h]);
  float ex = __expf(logits[t] - mx);
  logits[t] = ex;
  atomicAdd(sumbuf + d * 4 + h, ex);
}

// Edge pass 3: acc[dst] += xl[src] * (ex / (sum[dst] + 1e-16)).
template <int C>
__global__ void edge_agg_kernel(const int* __restrict__ srcI,
                                const int* __restrict__ dstI,
                                const float* __restrict__ xl,
                                const float* __restrict__ ex,
                                const float* __restrict__ sumbuf,
                                float* __restrict__ acc, int E) {
  const int HC = 4 * C, PER = HC / 32;
  int wave = threadIdx.x >> 5, lane = threadIdx.x & 31;
  int e = blockIdx.x * 8 + wave;
  if (e >= E) return;
  int s = srcI[e], d = dstI[e];
  int h = lane >> 3;
  float alpha = ex[(size_t)e * 4 + h] / (sumbuf[d * 4 + h] + 1e-16f);
  const float* pl = xl + (size_t)s * HC + lane * PER;
  float* pa = acc + (size_t)d * HC + lane * PER;
#pragma unroll
  for (int j = 0; j < PER; j++) atomicAdd(pa + j, pl[j] * alpha);
}

// conv2 epilogue: out[m,c] = mean_h acc[m, h*128 + c] + bias[c]
__global__ void mean_heads_bias_kernel(const float* __restrict__ acc,
                                       const float* __restrict__ bias,
                                       float* __restrict__ out, int M) {
  int t = blockIdx.x * 256 + threadIdx.x;
  if (t >= M * 128) return;
  int m = t >> 7, c = t & 127;
  const float* a = acc + (size_t)m * 512;
  out[t] = 0.25f * (a[c] + a[128 + c] + a[256 + c] + a[384 + c]) + bias[c];
}

// ---------------------------------------------------------------------------
// Host orchestration
// ---------------------------------------------------------------------------
extern "C" void kernel_launch(void* const* d_in, const int* in_sizes, int n_in,
                              void* d_out, int out_size, void* d_ws, size_t ws_size,
                              hipStream_t stream) {
  (void)n_in; (void)out_size; (void)ws_size;
  const float* x_token = (const float*)d_in[0];
  const float* x_pool  = (const float*)d_in[1];
  const int*   eidx    = (const int*)d_in[2];
  const float* Wt = (const float*)d_in[3],  *bt = (const float*)d_in[4];
  const float* Wp = (const float*)d_in[5],  *bp = (const float*)d_in[6];
  const float* g_tn  = (const float*)d_in[7],  *b_tn  = (const float*)d_in[8];
  const float* g_pn  = (const float*)d_in[9],  *b_pn  = (const float*)d_in[10];
  const float* g_htn = (const float*)d_in[11], *b_htn = (const float*)d_in[12];
  const float* g_hpn = (const float*)d_in[13], *b_hpn = (const float*)d_in[14];
  // c1_tp
  const float* W1t_l = (const float*)d_in[15], *b1t_l = (const float*)d_in[16];
  const float* W1t_r = (const float*)d_in[17], *b1t_r = (const float*)d_in[18];
  const float* att1t = (const float*)d_in[19], *bias1t = (const float*)d_in[20];
  // c1_pt
  const float* W1p_l = (const float*)d_in[21], *b1p_l = (const float*)d_in[22];
  const float* W1p_r = (const float*)d_in[23], *b1p_r = (const float*)d_in[24];
  const float* att1p = (const float*)d_in[25], *bias1p = (const float*)d_in[26];
  // c2_tp
  const float* W2t_l = (const float*)d_in[27], *b2t_l = (const float*)d_in[28];
  const float* W2t_r = (const float*)d_in[29], *b2t_r = (const float*)d_in[30];
  const float* att2t = (const float*)d_in[31], *bias2t = (const float*)d_in[32];
  // c2_pt
  const float* W2p_l = (const float*)d_in[33], *b2p_l = (const float*)d_in[34];
  const float* W2p_r = (const float*)d_in[35], *b2p_r = (const float*)d_in[36];
  const float* att2p = (const float*)d_in[37], *bias2p = (const float*)d_in[38];

  const int N_TOK  = in_sizes[0] / 64;
  const int N_POOL = in_sizes[1] / 32;
  const int E      = in_sizes[2] / 2;
  const int* srcI = eidx;           // token index per edge
  const int* dstI = eidx + E;       // pool  index per edge

  // ---------------- workspace layout (explicit reuse unions) ----------------
  char* base = (char*)d_ws;
  size_t off = 0;
  auto take = [&](size_t bytes) -> char* {
    char* p = base + off;
    off = (off + bytes + 255) & ~(size_t)255;
    return p;
  };
  const size_t TOK256 = (size_t)N_TOK * 256 * 4;   // 102.4 MB
  const size_t TOK512 = (size_t)N_TOK * 512 * 4;   // 204.8 MB
  const size_t POOL256 = (size_t)N_POOL * 256 * 4;
  const size_t POOL512 = (size_t)N_POOL * 512 * 4;

  char* R1 = take(TOK512);   // phase1/2: t | xl1      phase3: xl2
  char* R2 = take(TOK512);   // phase2: xr1p | acct1   phase3: xr2p
  char* R3 = take(TOK512);   // phase2/3: th (low)     phase3: acct2
  float* t_act  = (float*)R1;
  float* xl1    = (float*)(R1 + TOK256);
  float* xl2    = (float*)R1;
  float* xr1p   = (float*)R2;
  float* acct1  = (float*)(R2 + TOK256);
  float* xr2p   = (float*)R2;
  float* th     = (float*)R3;
  float* acct2  = (float*)R3;

  float* pl_act = (float*)take(POOL256);
  float* xr1    = (float*)take(POOL256);
  float* xl1p   = (float*)take(POOL256);
  float* accp1  = (float*)take(POOL256);
  float* ph     = (float*)take(POOL256);
  float* xr2    = (float*)take(POOL512);
  float* xl2p   = (float*)take(POOL512);
  float* accp2  = (float*)take(POOL512);

  float* logits1t = (float*)take((size_t)E * 4 * 4);
  float* logits1p = (float*)take((size_t)E * 4 * 4);
  float* logits2t = (float*)take((size_t)E * 4 * 4);
  float* logits2p = (float*)take((size_t)E * 4 * 4);
  int*   pmax1 = (int*)take((size_t)N_POOL * 4 * 4);
  float* psum1 = (float*)take((size_t)N_POOL * 4 * 4);
  int*   tmax1 = (int*)take((size_t)N_TOK * 4 * 4);
  float* tsum1 = (float*)take((size_t)N_TOK * 4 * 4);
  int*   pmax2 = (int*)take((size_t)N_POOL * 4 * 4);
  float* psum2 = (float*)take((size_t)N_POOL * 4 * 4);
  int*   tmax2 = (int*)take((size_t)N_TOK * 4 * 4);
  float* tsum2 = (float*)take((size_t)N_TOK * 4 * 4);

  unsigned short* pWt  = (unsigned short*)take(64 * 256 * 2);
  unsigned short* pWp  = (unsigned short*)take(32 * 256 * 2);
  unsigned short* pW1tl = (unsigned short*)take(256 * 256 * 2);
  unsigned short* pW1tr = (unsigned short*)take(256 * 256 * 2);
  unsigned short* pW1pl = (unsigned short*)take(256 * 256 * 2);
  unsigned short* pW1pr = (unsigned short*)take(256 * 256 * 2);
  unsigned short* pW2tl = (unsigned short*)take(256 * 512 * 2);
  unsigned short* pW2tr = (unsigned short*)take(256 * 512 * 2);
  unsigned short* pW2pl = (unsigned short*)take(256 * 512 * 2);
  unsigned short* pW2pr = (unsigned short*)take(256 * 512 * 2);

  float* outTok  = (float*)d_out;
  float* outPool = outTok + (size_t)N_TOK * 128;

  // ---------------- helpers ----------------
  auto fill = [&](void* p, unsigned v, size_t n) {
    fill_u32_kernel<<<(unsigned)((n + 255) / 256), 256, 0, stream>>>((unsigned*)p, v, n);
  };
  auto pack = [&](const float* W, unsigned short* P, int K, int N) {
    int total = (N / 16) * (K / 32) * 512;
    pack_w_kernel<<<(total + 255) / 256, 256, 0, stream>>>(W, P, K, N);
  };
  auto gemmb = [&](const float* A, const unsigned short* P, const float* bias,
                   float* C, int M, int N, int K) {
    dim3 g((M + 15) / 16, N / 256);
    gemm_bias_kernel<<<g, 256, 0, stream>>>(A, P, bias, C, M, N, K);
  };
  const unsigned NEG_INF = 0xFF800000u;
  const int eg = (E + 7) / 8;                 // wave-per-edge grids
  const int eg4 = (E * 4 + 255) / 256;

  // ---------------- 0. pack weights ----------------
  pack(Wt, pWt, 64, 256);       pack(Wp, pWp, 32, 256);
  pack(W1t_l, pW1tl, 256, 256); pack(W1t_r, pW1tr, 256, 256);
  pack(W1p_l, pW1pl, 256, 256); pack(W1p_r, pW1pr, 256, 256);
  pack(W2t_l, pW2tl, 256, 512); pack(W2t_r, pW2tr, 256, 512);
  pack(W2p_l, pW2pl, 256, 512); pack(W2p_r, pW2pr, 256, 512);

  // ---------------- 1. input projections + LN + ELU ----------------
  gemm_ln_elu_kernel<<<(N_TOK + 15) / 16, 256, 0, stream>>>(
      x_token, pWt, bt, g_tn, b_tn, t_act, N_TOK, 64);
  gemm_ln_elu_kernel<<<(N_POOL + 15) / 16, 256, 0, stream>>>(
      x_pool, pWp, bp, g_pn, b_pn, pl_act, N_POOL, 32);

  // ---------------- 2. conv1 projections ----------------
  gemmb(t_act,  pW1tl, b1t_l, xl1,  N_TOK,  256, 256);  // c1_tp: xl = t @ Wl
  gemmb(pl_act, pW1tr, b1t_r, xr1,  N_POOL, 256, 256);  // c1_tp: xr = pl @ Wr
  gemmb(pl_act, pW1pl, b1p_l, xl1p, N_POOL, 256, 256);  // c1_pt: xl = pl @ Wl
  gemmb(t_act,  pW1pr, b1p_r, xr1p, N_TOK,  256, 256);  // c1_pt: xr = t @ Wr

  // ---------------- 3. conv1 edge softmax + aggregation ----------------
  fill(pmax1, NEG_INF, (size_t)N_POOL * 4);
  fill(psum1, 0u, (size_t)N_POOL * 4);
  fill(tmax1, NEG_INF, (size_t)N_TOK * 4);
  fill(tsum1, 0u, (size_t)N_TOK * 4);
  fill(accp1, 0u, (size_t)N_POOL * 256);
  fill(acct1, 0u, (size_t)N_TOK * 256);

  edge_logits_kernel<64><<<eg, 256, 0, stream>>>(srcI, dstI, xl1,  xr1,  att1t, logits1t, pmax1, E);
  edge_logits_kernel<64><<<eg, 256, 0, stream>>>(dstI, srcI, xl1p, xr1p, att1p, logits1p, tmax1, E);
  edge_exp_kernel<<<eg4, 256, 0, stream>>>(dstI, logits1t, pmax1, psum1, E);
  edge_exp_kernel<<<eg4, 256, 0, stream>>>(srcI, logits1p, tmax1, tsum1, E);
  edge_agg_kernel<64><<<eg, 256, 0, stream>>>(srcI, dstI, xl1,  logits1t, psum1, accp1, E);
  edge_agg_kernel<64><<<eg, 256, 0, stream>>>(dstI, srcI, xl1p, logits1p, tsum1, acct1, E);

  // ---------------- 4. hidden LN + ELU ----------------
  bias_ln_elu_kernel<<<(N_TOK + 7) / 8, 256, 0, stream>>>(acct1, bias1p, g_htn, b_htn, th, N_TOK);
  bias_ln_elu_kernel<<<(N_POOL + 7) / 8, 256, 0, stream>>>(accp1, bias1t, g_hpn, b_hpn, ph, N_POOL);

  // ---------------- 5. conv2 projections ----------------
  gemmb(th, pW2tl, b2t_l, xl2,  N_TOK,  512, 256);      // c2_tp: xl = th @ Wl
  gemmb(ph, pW2tr, b2t_r, xr2,  N_POOL, 512, 256);      // c2_tp: xr = ph @ Wr
  gemmb(ph, pW2pl, b2p_l, xl2p, N_POOL, 512, 256);      // c2_pt: xl = ph @ Wl
  gemmb(th, pW2pr, b2p_r, xr2p, N_TOK,  512, 256);      // c2_pt: xr = th @ Wr

  // ---------------- 6. conv2 edge softmax + aggregation ----------------
  fill(pmax2, NEG_INF, (size_t)N_POOL * 4);
  fill(psum2, 0u, (size_t)N_POOL * 4);
  fill(tmax2, NEG_INF, (size_t)N_TOK * 4);
  fill(tsum2, 0u, (size_t)N_TOK * 4);
  fill(accp2, 0u, (size_t)N_POOL * 512);
  fill(acct2, 0u, (size_t)N_TOK * 512);   // (aliases th: th is dead now)

  edge_logits_kernel<128><<<eg, 256, 0, stream>>>(srcI, dstI, xl2,  xr2,  att2t, logits2t, pmax2, E);
  edge_logits_kernel<128><<<eg, 256, 0, stream>>>(dstI, srcI, xl2p, xr2p, att2p, logits2p, tmax2, E);
  edge_exp_kernel<<<eg4, 256, 0, stream>>>(dstI, logits2t, pmax2, psum2, E);
  edge_exp_kernel<<<eg4, 256, 0, stream>>>(srcI, logits2p, tmax2, tsum2, E);
  edge_agg_kernel<128><<<eg, 256, 0, stream>>>(srcI, dstI, xl2,  logits2t, psum2, accp2, E);
  edge_agg_kernel<128><<<eg, 256, 0, stream>>>(dstI, srcI, xl2p, logits2p, tsum2, acct2, E);

  // ---------------- 7. head-mean + bias -> outputs ----------------
  mean_heads_bias_kernel<<<(N_TOK * 128 + 255) / 256, 256, 0, stream>>>(acct2, bias2p, outTok, N_TOK);
  mean_heads_bias_kernel<<<(N_POOL * 128 + 255) / 256, 256, 0, stream>>>(accp2, bias2t, outPool, N_POOL);
}
